// Ant_Model_51659866637021
// MI455X (gfx1250) — matrix operand order repaired and checked by reading, not compiled, not run on hardware
//
#include <hip/hip_runtime.h>

typedef __bf16 bf16_t;
typedef __attribute__((ext_vector_type(16))) __bf16 v16bf;
typedef __attribute__((ext_vector_type(8)))  __bf16 v8bf;
typedef __attribute__((ext_vector_type(8)))  float  v8f;
typedef __attribute__((ext_vector_type(8)))  float  f32x8;

// global-address-space vector load helpers (force global_load_b128 codegen)
typedef const f32x8 __attribute__((address_space(1)))* f32x8_gptr;
typedef const v8bf  __attribute__((address_space(1)))* v8bf_gptr;

__device__ __forceinline__ f32x8 gload8f(const float* p) {
    return *(f32x8_gptr)p;
}
__device__ __forceinline__ v8bf gload8bf(const bf16_t* p) {
    return *(v8bf_gptr)p;
}

namespace {
constexpr int T = 32, B = 32, F = 1024, H = 1024;
constexpr int VC = 125, NC = 352, CC = 2513;
constexpr int PTOT = T * (T - 1) / 2;   // 496
constexpr int G3H = 3 * H;              // 3072
constexpr float EPSV = 1e-6f;
}

// v_cvt_pk_bf16_f32: pack two f32 -> {bf16(hi)[31:16], bf16(lo)[15:0]}
__device__ __forceinline__ unsigned cvt_pk_bf16(float lo, float hi) {
    unsigned r;
    asm("v_cvt_pk_bf16_f32 %0, %1, %2" : "=v"(r) : "v"(lo), "v"(hi));
    return r;
}

// v_pk_max_num_bf16 with +0: packed-pair ReLU on two bf16 lanes (1 op / 2 elems)
__device__ __forceinline__ unsigned pk_relu_bf16(unsigned a) {
    unsigned r;
    asm("v_pk_max_num_bf16 %0, %1, 0" : "=v"(r) : "v"(a));
    return r;
}

// Assemble a 16-element bf16 fragment from two 8-float segments using packed
// conversion (element pair (2i,2i+1) occupies one 32-bit VGPR); optional
// fused ReLU applied on the packed pairs.
template <bool RELU>
__device__ __forceinline__ v16bf cvt16(const f32x8 lo, const f32x8 hi) {
    union { v16bf v; unsigned u[8]; } r;
#pragma unroll
    for (int i = 0; i < 4; ++i) {
        unsigned a = cvt_pk_bf16(lo[2 * i], lo[2 * i + 1]);
        r.u[i] = RELU ? pk_relu_bf16(a) : a;
    }
#pragma unroll
    for (int i = 0; i < 4; ++i) {
        unsigned a = cvt_pk_bf16(hi[2 * i], hi[2 * i + 1]);
        r.u[4 + i] = RELU ? pk_relu_bf16(a) : a;
    }
    return r.v;
}

// ---------------------------------------------------------------------------
// Batched strided GEMM with split-K A operand:
//   C[b] = act(concatK(A[b], A2[b])) * B[b]^T (+ bias) (+= C)
// A/A2: f32, vector-loaded, packed-converted to bf16 with fused packed ReLU.
// B: bf16; BKC=true -> ldb_k==1, vectorized 16B fragment loads.
// 8 waves/block (2x4), wave tile 32x64 (2x4 WMMA fragments), block 64M x 256N.
// OOB rows/cols are clamped on load (results land in never-stored lanes).
// ---------------------------------------------------------------------------
struct GemmParams {
    const float*  A;   int lda;            long long batchA;
    const float*  A2;  int lda2;           // used for k >= Ksplit (batch stride 0)
    const bf16_t* Bm;  int ldb_n, ldb_k;   long long batchB;
    float*        C;   int ldc;            long long batchC;
    const float*  bias;                    // nullable, indexed by output column
    int M, N, K, Ksplit;
    int flags;                             // bit1: accumulate into C
};

template <bool BKC, bool RELU>
__global__ __launch_bounds__(256) void gemm_bf16_wmma(GemmParams p) {
    const int bz = blockIdx.z;
    const float*  A  = p.A  + (long long)bz * p.batchA;
    const bf16_t* Bm = p.Bm + (long long)bz * p.batchB;
    float*        C  = p.C  + (long long)bz * p.batchC;

    const int wid    = threadIdx.x >> 5;
    const int lane   = threadIdx.x & 31;
    const int lr     = lane & 15;       // row-in-fragment / col-in-fragment
    const int lk     = lane >> 4;       // half-wave selector
    const int wave_m = wid >> 2;        // 0..1
    const int wave_n = wid & 3;         // 0..3
    const int m0 = blockIdx.y * 64 + wave_m * 32;
    const int n0 = blockIdx.x * 256 + wave_n * 64;

    // per-fragment A row pointers (clamped rows; advance by 32 per k-chunk)
    int rowc[2];
    const float* apk[2];
#pragma unroll
    for (int mi = 0; mi < 2; ++mi) {
        const int row = m0 + mi * 16 + lr;
        rowc[mi] = (row < p.M) ? row : (p.M - 1);
        apk[mi] = A + (long long)rowc[mi] * p.lda;
    }

    // per-fragment B col pointers (clamped cols; pre-offset by lk*16)
    const bf16_t* bpk[4];
#pragma unroll
    for (int ni = 0; ni < 4; ++ni) {
        const int col = n0 + ni * 16 + lr;
        const int colc = (col < p.N) ? col : (p.N - 1);
        bpk[ni] = Bm + (long long)colc * p.ldb_n
                     + (BKC ? (long long)(lk * 16)
                            : (long long)(lk * 16) * p.ldb_k);
    }

    v8f acc[2][4];
#pragma unroll
    for (int i = 0; i < 2; ++i)
#pragma unroll
        for (int j = 0; j < 4; ++j)
#pragma unroll
            for (int r = 0; r < 8; ++r) acc[i][j][r] = 0.0f;

    for (int k0 = 0; k0 < p.K; k0 += 32) {
        // uniform one-time switch to second A segment (virtual K-concat)
        if (k0 == p.Ksplit) {
#pragma unroll
            for (int mi = 0; mi < 2; ++mi)
                apk[mi] = p.A2 + (long long)rowc[mi] * p.lda2;
        }

        // ---- A fragments: 16x32 bf16, lane lr = row, K 0-15 / 16-31
        v16bf afrag[2];
#pragma unroll
        for (int mi = 0; mi < 2; ++mi) {
            f32x8 flo = gload8f(apk[mi] + lk * 16);       // K seg 0
            f32x8 fhi = gload8f(apk[mi] + lk * 16 + 8);   // K seg 1
            afrag[mi] = cvt16<RELU>(flo, fhi);
            apk[mi] += 32;
        }

        // ---- B fragments: 32x16 bf16, lane lr = output col, K 0-15 / 16-31
        v16bf bfrag[4];
#pragma unroll
        for (int ni = 0; ni < 4; ++ni) {
            if (BKC) {
                v8bf lo = gload8bf(bpk[ni]);
                v8bf hi = gload8bf(bpk[ni] + 8);
                bfrag[ni] = __builtin_shufflevector(lo, hi, 0, 1, 2, 3, 4, 5, 6, 7,
                                                    8, 9, 10, 11, 12, 13, 14, 15);
                bpk[ni] += 32;
            } else {
                v16bf bfv;
#pragma unroll
                for (int e = 0; e < 16; ++e) bfv[e] = bpk[ni][(long long)e * p.ldb_k];
                bfrag[ni] = bfv;
                bpk[ni] += (long long)32 * p.ldb_k;
            }
        }

#pragma unroll
        for (int mi = 0; mi < 2; ++mi)
#pragma unroll
            for (int ni = 0; ni < 4; ++ni)
                acc[mi][ni] = __builtin_amdgcn_wmma_f32_16x16x32_bf16(
                    false, afrag[mi], false, bfrag[ni],
                    (short)0, acc[mi][ni], false, false);
    }

    // ---- store: lane lr = col, rows = lk*8 + r (real indices, guarded)
#pragma unroll
    for (int mi = 0; mi < 2; ++mi) {
#pragma unroll
        for (int ni = 0; ni < 4; ++ni) {
            const int col = n0 + ni * 16 + lr;
            if (col >= p.N) continue;
            const float badd = p.bias ? p.bias[col] : 0.0f;
#pragma unroll
            for (int r = 0; r < 8; ++r) {
                const int row = m0 + mi * 16 + lk * 8 + r;
                if (row >= p.M) continue;
                const long long ci = (long long)row * p.ldc + col;
                float v = acc[mi][ni][r] + badd;
                if (p.flags & 2) v += C[ci];
                C[ci] = v;
            }
        }
    }
}

// ---------------------------------------------------------------------------
// Pointwise / utility kernels
// ---------------------------------------------------------------------------
__global__ void cvt_f32_bf16(const float* __restrict__ src,
                             bf16_t* __restrict__ dst, long long n) {
    long long i = (long long)blockIdx.x * blockDim.x + threadIdx.x;
    if (i < n) dst[i] = (bf16_t)src[i];
}

// inputs [B,T,F] -> x [T,B,F] (f32) and x_bf (bf16)
__global__ void transpose_x(const float* __restrict__ in,
                            float* __restrict__ x, bf16_t* __restrict__ xbf) {
    long long i = (long long)blockIdx.x * blockDim.x + threadIdx.x;
    if (i >= (long long)T * B * F) return;
    int f = (int)(i % F);
    long long r = i / F;
    int bb = (int)(r % B);
    int t  = (int)(r / B);
    float v = in[(long long)bb * T * F + (long long)t * F + f];
    x[i]   = v;
    xbf[i] = (bf16_t)v;
}

// one wave per row: out[row] = ||src[row,:]||_2
__global__ void row_norm(const float* __restrict__ src, float* __restrict__ out,
                         int rows, int L) {
    int row  = blockIdx.x * (blockDim.x >> 5) + (threadIdx.x >> 5);
    int lane = threadIdx.x & 31;
    if (row >= rows) return;
    const float* p = src + (long long)row * L;
    float s = 0.0f;
    for (int i = lane; i < L; i += 32) { float v = p[i]; s += v * v; }
#pragma unroll
    for (int off = 16; off > 0; off >>= 1) s += __shfl_xor(s, off, 32);
    if (lane == 0) out[row] = sqrtf(s);
}

// GRU gate fusion (gate order r, z, n); hout may alias h
__global__ void gru_pointwise(const float* __restrict__ gi,
                              const float* __restrict__ gh,
                              const float* __restrict__ h,
                              float* __restrict__ hout, int rows) {
    long long i = (long long)blockIdx.x * blockDim.x + threadIdx.x;
    if (i >= (long long)rows * H) return;
    long long row = i / H;
    int col = (int)(i % H);
    long long g = row * (long long)G3H + col;
    float r = 1.0f / (1.0f + __expf(-(gi[g] + gh[g])));
    float z = 1.0f / (1.0f + __expf(-(gi[g + H] + gh[g + H])));
    float n = tanhf(gi[g + 2 * H] + r * gh[g + 2 * H]);
    hout[i] = (1.0f - z) * n + z * h[i];
}

// sim[t,b,j] = dots/(max(nq*nk,eps)) * tril mask
__global__ void sim_kernel(const float* __restrict__ dots,
                           const float* __restrict__ nq,
                           const float* __restrict__ nk,
                           float* __restrict__ sim) {
    int i = blockIdx.x * blockDim.x + threadIdx.x;
    if (i >= T * B * T) return;
    int j  = i % T;
    int r  = i / T;
    int bb = r % B;
    int t  = r / B;
    float den = nq[t * B + bb] * nk[j * B + bb];
    den = den > EPSV ? den : EPSV;
    sim[i] = (j <= t) ? dots[i] / den : 0.0f;
}

// fp[b, off(t)+s, :] = h_t[t,b,:]   for t in [0, nvalid)
__global__ void fp_record(const float* __restrict__ ht, float* __restrict__ fp,
                          int s, int nvalid) {
    long long i = (long long)blockIdx.x * blockDim.x + threadIdx.x;
    if (i >= (long long)nvalid * B * H) return;
    int col = (int)(i % H);
    long long r = i / H;
    int bb = (int)(r % B);
    int t  = (int)(r / B);
    int off = t * (T - 1) - t * (t - 1) / 2;
    fp[((long long)bb * PTOT + off + s) * H + col] =
        ht[((long long)t * B + bb) * H + col];
}

// clc[b, tstar, :] = concat(h_t[tstar,b,:], h_re[tstar,b,:])
__global__ void clc_record(const float* __restrict__ ht,
                           const float* __restrict__ hre,
                           float* __restrict__ clc, int tstar) {
    int i = blockIdx.x * blockDim.x + threadIdx.x;
    if (i >= B * 2 * H) return;
    int col = i % (2 * H);
    int bb  = i / (2 * H);
    long long src = ((long long)tstar * B + bb) * H;
    float v = (col < H) ? ht[src + col] : hre[src + col - H];
    clc[((long long)bb * T + tstar) * (2 * H) + col] = v;
}

// ---------------------------------------------------------------------------
// Host orchestration
// ---------------------------------------------------------------------------
static inline dim3 ew_grid(long long n) { return dim3((unsigned)((n + 255) / 256)); }

static void launch_gemm_split(hipStream_t s,
                              const float* A, int lda, long long bA,
                              const float* A2, int lda2, int Ksplit,
                              const bf16_t* Bm, int ldbn, int ldbk, long long bB,
                              float* C, int ldc, long long bC,
                              const float* bias, int M, int N, int K,
                              int batches, int flags) {
    GemmParams p;
    p.A = A; p.lda = lda; p.batchA = bA;
    p.A2 = A2 ? A2 : A; p.lda2 = A2 ? lda2 : lda;
    p.Bm = Bm; p.ldb_n = ldbn; p.ldb_k = ldbk; p.batchB = bB;
    p.C = C; p.ldc = ldc; p.batchC = bC;
    p.bias = bias; p.M = M; p.N = N; p.K = K;
    p.Ksplit = A2 ? Ksplit : K;
    p.flags = flags;
    dim3 grid((N + 255) / 256, (M + 63) / 64, batches);
    const bool relu = (flags & 1) != 0;
    if (ldbk == 1) {
        if (relu) hipLaunchKernelGGL((gemm_bf16_wmma<true, true>),  grid, dim3(256), 0, s, p);
        else      hipLaunchKernelGGL((gemm_bf16_wmma<true, false>), grid, dim3(256), 0, s, p);
    } else {
        if (relu) hipLaunchKernelGGL((gemm_bf16_wmma<false, true>),  grid, dim3(256), 0, s, p);
        else      hipLaunchKernelGGL((gemm_bf16_wmma<false, false>), grid, dim3(256), 0, s, p);
    }
}

static void launch_gemm(hipStream_t s,
                        const float* A, int lda, long long bA,
                        const bf16_t* Bm, int ldbn, int ldbk, long long bB,
                        float* C, int ldc, long long bC,
                        const float* bias, int M, int N, int K, int batches,
                        int flags) {
    launch_gemm_split(s, A, lda, bA, nullptr, 0, K, Bm, ldbn, ldbk, bB,
                      C, ldc, bC, bias, M, N, K, batches, flags);
}

extern "C" void kernel_launch(void* const* d_in, const int* in_sizes, int n_in,
                              void* d_out, int out_size, void* d_ws, size_t ws_size,
                              hipStream_t stream) {
    const float* inputs   = (const float*)d_in[0];
    const float* enc_Wih  = (const float*)d_in[1];
    const float* enc_Whh  = (const float*)d_in[2];
    const float* enc_bih  = (const float*)d_in[3];
    const float* enc_bhh  = (const float*)d_in[4];
    const float* ant_Wih  = (const float*)d_in[5];
    const float* ant_Whh  = (const float*)d_in[6];
    const float* ant_bih  = (const float*)d_in[7];
    const float* ant_bhh  = (const float*)d_in[8];
    const float* re_Wih   = (const float*)d_in[9];
    const float* re_Whh   = (const float*)d_in[10];
    const float* re_bih   = (const float*)d_in[11];
    const float* re_bhh   = (const float*)d_in[12];
    const float* Wv = (const float*)d_in[13]; const float* bv = (const float*)d_in[14];
    const float* Wn = (const float*)d_in[15]; const float* bn = (const float*)d_in[16];
    const float* Wc = (const float*)d_in[17]; const float* bc = (const float*)d_in[18];

    float* out      = (float*)d_out;
    float* y_out    = out;                                   // [B,T,CC]
    float* fp_out   = y_out + (size_t)B * T * CC;            // [B,PTOT,H]
    float* verb_out = fp_out + (size_t)B * PTOT * H;         // [B,T,VC]
    float* noun_out = verb_out + (size_t)B * T * VC;         // [B,T,NC]

    // ---- workspace layout (256B aligned slabs) ----
    char* wp = (char*)d_ws;
    auto alloc = [&](size_t bytes) -> char* {
        char* r = wp; wp += (bytes + 255) & ~(size_t)255; return r;
    };
    bf16_t* encWih_b = (bf16_t*)alloc((size_t)G3H * F * 2);
    bf16_t* encWhh_b = (bf16_t*)alloc((size_t)G3H * H * 2);
    bf16_t* antWih_b = (bf16_t*)alloc((size_t)G3H * (H + F) * 2);
    bf16_t* antWhh_b = (bf16_t*)alloc((size_t)G3H * H * 2);
    bf16_t* reWih_b  = (bf16_t*)alloc((size_t)G3H * (F + H) * 2);
    bf16_t* reWhh_b  = (bf16_t*)alloc((size_t)G3H * H * 2);
    bf16_t* Wv_b     = (bf16_t*)alloc((size_t)VC * 2 * H * 2);
    bf16_t* Wn_b     = (bf16_t*)alloc((size_t)NC * 2 * H * 2);
    bf16_t* Wc_b     = (bf16_t*)alloc((size_t)CC * 2 * H * 2);

    float*  x     = (float*)alloc((size_t)T * B * F * 4);
    bf16_t* x_bf  = (bf16_t*)alloc((size_t)T * B * F * 2);
    float*  feats = (float*)alloc((size_t)T * B * H * 4);
    float*  h_t   = (float*)alloc((size_t)T * B * H * 4);
    float*  h_re  = (float*)alloc((size_t)T * B * H * 4);
    float*  gi    = (float*)alloc((size_t)T * B * G3H * 4);
    float*  gh    = (float*)alloc((size_t)T * B * G3H * 4);
    float*  reat  = (float*)alloc((size_t)T * B * F * 4);
    float*  dots  = (float*)alloc((size_t)T * B * T * 4);
    float*  sim   = (float*)alloc((size_t)T * B * T * 4);
    float*  nk    = (float*)alloc((size_t)T * B * 4);
    float*  nq    = (float*)alloc((size_t)T * B * 4);
    float*  clc   = (float*)alloc((size_t)B * T * 2 * H * 4);
    float*  zbuf  = (float*)alloc((size_t)B * H * 4);
    (void)ws_size; (void)n_in; (void)in_sizes; (void)out_size;

    hipMemsetAsync(zbuf, 0, (size_t)B * H * 4, stream);

    // ---- one-time conversions ----
    auto cvt = [&](const float* s, bf16_t* d, long long n) {
        hipLaunchKernelGGL(cvt_f32_bf16, ew_grid(n), dim3(256), 0, stream, s, d, n);
    };
    cvt(enc_Wih, encWih_b, (long long)G3H * F);
    cvt(enc_Whh, encWhh_b, (long long)G3H * H);
    cvt(ant_Wih, antWih_b, (long long)G3H * (H + F));
    cvt(ant_Whh, antWhh_b, (long long)G3H * H);
    cvt(re_Wih,  reWih_b,  (long long)G3H * (F + H));
    cvt(re_Whh,  reWhh_b,  (long long)G3H * H);
    cvt(Wv, Wv_b, (long long)VC * 2 * H);
    cvt(Wn, Wn_b, (long long)NC * 2 * H);
    cvt(Wc, Wc_b, (long long)CC * 2 * H);

    hipLaunchKernelGGL(transpose_x, ew_grid((long long)T * B * F), dim3(256), 0,
                       stream, inputs, x, x_bf);
    hipLaunchKernelGGL(row_norm, dim3((T * B + 7) / 8), dim3(256), 0, stream,
                       x, nk, T * B, F);

    // ---- encoder scan: feats[t] = GRU(relu(x[t]), feats[t-1]) ----
    for (int t = 0; t < T; ++t) {
        const float* xt = x + (long long)t * B * F;
        const float* hprev = (t == 0) ? zbuf : feats + (long long)(t - 1) * B * H;
        launch_gemm(stream, xt, F, 0, encWih_b, F, 1, 0, gi, G3H, 0,
                    enc_bih, B, G3H, F, 1, /*relu_a*/1);
        launch_gemm(stream, hprev, H, 0, encWhh_b, H, 1, 0, gh, G3H, 0,
                    enc_bhh, B, G3H, H, 1, 0);
        hipLaunchKernelGGL(gru_pointwise, ew_grid((long long)B * H), dim3(256), 0,
                           stream, gi, gh, hprev, feats + (long long)t * B * H, B);
    }

    // ---- init ant-scan carries ----
    hipMemcpyAsync(h_t,  feats, (size_t)T * B * H * 4, hipMemcpyDeviceToDevice, stream);
    hipMemcpyAsync(h_re, feats, (size_t)T * B * H * 4, hipMemcpyDeviceToDevice, stream);

    const int RTB = T * B;  // 1024 rows
    for (int s = 0; s < T; ++s) {
        // ant gi = relu(concat(h_re|x, x)) @ ant_Wih^T + bih  (split-K fused)
        const float* a1 = (s == 0) ? x : h_re;
        int lda1 = (s == 0) ? F : H;
        launch_gemm_split(stream, a1, lda1, 0, x, F, H,
                          antWih_b, H + F, 1, 0, gi, G3H, 0,
                          ant_bih, RTB, G3H, H + F, 1, /*relu*/1);
        // ant gh = h_t @ ant_Whh^T + bhh
        launch_gemm(stream, h_t, H, 0, antWhh_b, H, 1, 0, gh, G3H, 0,
                    ant_bhh, RTB, G3H, H, 1, 0);
        hipLaunchKernelGGL(gru_pointwise, ew_grid((long long)RTB * H), dim3(256), 0,
                           stream, gi, gh, h_t, h_t, RTB);   // q = h_t (updated)

        // record feature predictions for valid t
        int nvalid = T - 1 - s;
        if (nvalid > 0)
            hipLaunchKernelGGL(fp_record, ew_grid((long long)nvalid * B * H),
                               dim3(256), 0, stream, h_t, fp_out, s, nvalid);

        // attention: dots[t,b,j] = q[t,b,:].x[j,b,:] ; batched over b
        hipLaunchKernelGGL(row_norm, dim3((T * B + 7) / 8), dim3(256), 0, stream,
                           h_t, nq, T * B, H);
        launch_gemm(stream, h_t, B * H, H,
                    x_bf, B * F, 1, F,
                    dots, B * T, T,
                    nullptr, T, T, H, B, 0);
        hipLaunchKernelGGL(sim_kernel, ew_grid(T * B * T), dim3(256), 0, stream,
                           dots, nq, nk, sim);
        // reat[t,b,f] = sum_j sim[t,b,j] * x[j,b,f]  (strided-K B operand)
        launch_gemm(stream, sim, B * T, T,
                    x_bf, 1, B * F, F,
                    reat, B * F, F,
                    nullptr, T, F, T, B, 0);

        // re gi = relu(concat(q, reat)) @ re_Wih^T + bih  (split-K fused)
        launch_gemm_split(stream, h_t, H, 0, reat, F, H,
                          reWih_b, F + H, 1, 0, gi, G3H, 0,
                          re_bih, RTB, G3H, F + H, 1, 1);
        // re gh = h_re @ re_Whh^T + bhh
        launch_gemm(stream, h_re, H, 0, reWhh_b, H, 1, 0, gh, G3H, 0,
                    re_bhh, RTB, G3H, H, 1, 0);
        hipLaunchKernelGGL(gru_pointwise, ew_grid((long long)RTB * H), dim3(256), 0,
                           stream, gi, gh, h_re, h_re, RTB);

        // clc kept for t* = T-1-s
        hipLaunchKernelGGL(clc_record, ew_grid(B * 2 * H), dim3(256), 0, stream,
                           h_t, h_re, clc, T - 1 - s);
    }

    // ---- classifiers: rf = relu(clc) ; rows m = b*T+t ----
    launch_gemm(stream, clc, 2 * H, 0, Wc_b, 2 * H, 1, 0, y_out, CC, 0,
                bc, B * T, CC, 2 * H, 1, 1);
    launch_gemm(stream, clc, 2 * H, 0, Wv_b, 2 * H, 1, 0, verb_out, VC, 0,
                bv, B * T, VC, 2 * H, 1, 1);
    launch_gemm(stream, clc, 2 * H, 0, Wn_b, 2 * H, 1, 0, noun_out, NC, 0,
                bn, B * T, NC, 2 * H, 1, 1);
}
